// FoundationDepthLSS_45981919871117
// MI455X (gfx1250) — matrix operand order repaired and checked by reading, not compiled, not run on hardware
//
#include <hip/hip_runtime.h>

#define FH 32
#define FW 88
#define HW (FH*FW)          // 2816
#define BNC 6
#define NPIX (BNC*HW)       // 16896
#define NDEPTH 59
#define CFEAT 128
#define XY 16384            // 128*128 BEV cells

typedef __attribute__((ext_vector_type(16))) _Float16 v16h;
typedef __attribute__((ext_vector_type(8)))  float    v8f;

union Frag {
  uint4 q[2];
  unsigned int u[8];
  v16h v;
};

// ---------------------------------------------------------------------------
// Pack OIHW f32 weights into per-lane WMMA A-fragment order (f16), K = ci*KS*KS + r.
// Fragment index: (((coTile*KB + kb)*32 + lane)*8 + vgpr) -> u32 of two f16 (k, k+1).
// A-layout (16-bit 16x32): lane<16 -> K base {2v | v<4 ; 16+2(v-4)}, lane>=16 -> +8.
// ---------------------------------------------------------------------------
template<int CIN, int COUT, int KS>
__global__ __launch_bounds__(256) void pack_weights(const float* __restrict__ w,
                                                    unsigned int* __restrict__ wp) {
  constexpr int KSS = KS*KS;
  constexpr int KB  = (CIN*KSS)/32;
  constexpr int TOT = (COUT/16)*KB*256;
  int e = blockIdx.x*256 + threadIdx.x;
  if (e >= TOT) return;
  int v    = e & 7;
  int lane = (e >> 3) & 31;
  int rem  = e >> 8;            // coTile*KB + kb
  int kb   = rem % KB;
  int t    = rem / KB;
  int co   = t*16 + (lane & 15);
  int khalf = (lane & 16) ? 8 : 0;
  int kbase = ((v < 4) ? (2*v) : (8 + 2*v)) + khalf;
  int k0 = kb*32 + kbase;
  union { _Float16 h[2]; unsigned int u; } pk;
#pragma unroll
  for (int j = 0; j < 2; ++j) {
    int k  = k0 + j;
    int ci = k / KSS;
    int r  = k - ci*KSS;
    pk.h[j] = (_Float16)w[(size_t)(co*CIN + ci)*KSS + r];
  }
  wp[e] = pk.u;
}

// ---------------------------------------------------------------------------
// Fold conv bias + eval-mode BN into per-channel scale/bias; copy conv3 bias.
// ---------------------------------------------------------------------------
__global__ __launch_bounds__(256) void prep_sb(
    const float* __restrict__ c1b, const float* __restrict__ g1, const float* __restrict__ bb1,
    const float* __restrict__ m1,  const float* __restrict__ v1,
    const float* __restrict__ c2b, const float* __restrict__ g2, const float* __restrict__ bb2,
    const float* __restrict__ m2,  const float* __restrict__ v2,
    const float* __restrict__ c3b,
    float* __restrict__ sb1, float* __restrict__ sb2, float* __restrict__ b3) {
  int t = blockIdx.x*256 + threadIdx.x;
  if (t < 256) {
    float inv = g1[t] * rsqrtf(v1[t] + 1e-3f);
    sb1[t] = inv;
    sb1[256 + t] = (c1b[t] - m1[t]) * inv + bb1[t];
  } else if (t < 512) {
    int i = t - 256;
    float inv = g2[i] * rsqrtf(v2[i] + 1e-3f);
    sb2[i] = inv;
    sb2[256 + i] = (c2b[i] - m2[i]) * inv + bb2[i];
  } else if (t < 640) {
    b3[t - 512] = c3b[t - 512];
  }
}

// ---------------------------------------------------------------------------
// Depth aligner + Laplace-kernel softmax over 59 bins (bins = 1..59).
// dist layout: (BN, D, H, W)
// ---------------------------------------------------------------------------
__global__ __launch_bounds__(256) void depth_kernel(
    const float* __restrict__ rel, const float* __restrict__ log_scale,
    const float* __restrict__ shift, const float* __restrict__ log_sigma,
    float* __restrict__ dist) {
  int p = blockIdx.x*256 + threadIdx.x;
  if (p >= NPIX) return;
  float sc  = expf(log_scale[0]);
  float m   = fminf(fmaxf(sc * rel[p] + shift[0], 0.5f), 150.0f);
  float sig = fminf(fmaxf(expf(log_sigma[0]), 0.1f), 20.0f);
  float inv = 1.0f / sig;
  float amin = 1e30f;
  for (int d = 0; d < NDEPTH; ++d) {
    float a = fabsf(m - (1.0f + (float)d));
    amin = fminf(amin, a);
  }
  float sum = 0.0f;
  for (int d = 0; d < NDEPTH; ++d) {
    float a = fabsf(m - (1.0f + (float)d));
    sum += expf((amin - a) * inv);
  }
  float rs = 1.0f / sum;
  int bn = p / HW, hw = p - bn*HW;
  float* dp = dist + (size_t)bn*NDEPTH*HW + hw;
  for (int d = 0; d < NDEPTH; ++d) {
    float a = fabsf(m - (1.0f + (float)d));
    dp[(size_t)d*HW] = expf((amin - a) * inv) * rs;
  }
}

// ---------------------------------------------------------------------------
// Implicit-GEMM conv via v_wmma_f32_16x16x32_f16.
// Block = 128 threads (4 waves). grid.x = pixel tiles (16 px, never cross a
// camera: HW=2816 is a multiple of 16). Each wave owns TWO 16-row co tiles.
// K is staged 64 at a time into a shared LDS B tile -> 4 WMMAs per wave per
// barrier pair. Pixel coords hoisted out of the K loop (no div in hot loop).
// Weight fragments for the next K step are prefetched (global_prefetch).
// MODE 0: NCHW out + fused BN/ReLU.  MODE 1: NHWC out + bias only.
// ---------------------------------------------------------------------------
template<int CIN, int COUT, int KS, int PAD, int MODE>
__global__ __launch_bounds__(128) void conv_wmma(
    const float* __restrict__ in, const unsigned int* __restrict__ wpack,
    const float* __restrict__ scale, const float* __restrict__ bias,
    float* __restrict__ out) {
  constexpr int KSS = KS*KS;
  constexpr int KK  = CIN*KSS;
  constexpr int KB2 = KK/64;              // 64-wide K steps
  __shared__ _Float16 Bs[16*66];          // [n][k0..63], pitch 66 -> conflict-free

  const int tid  = threadIdx.x;
  const int lane = tid & 31;
  const int wave = tid >> 5;
  const int coT0 = (blockIdx.y*4 + wave)*2;
  const int P0   = blockIdx.x*16;
  const int bn   = P0 / HW;
  const int hw0  = P0 - bn*HW;

  // staging: 1024 f16 elems / 128 threads = 8 each (fixed k, 8 consecutive n)
  const int kl = tid >> 1;                // 0..63
  const int n0 = (tid & 1)*8;

  // K-invariant pixel coordinates for this thread's 8 pixels (hoisted divides)
  int ph[8], pw[8];
#pragma unroll
  for (int j = 0; j < 8; ++j) {
    int hw = hw0 + n0 + j;
    ph[j] = hw / FW;
    pw[j] = hw - ph[j]*FW;
  }

  v8f acc[2];
  acc[0] = {};
  acc[1] = {};

  // fragment stream bases; each 64-K step consumes 128 uint4 per co tile
  const uint4* wpA0 = (const uint4*)(wpack + ((size_t)coT0     * (KK/32)*32 + lane)*8);
  const uint4* wpA1 = (const uint4*)(wpack + ((size_t)(coT0+1) * (KK/32)*32 + lane)*8);

  for (int kb2 = 0; kb2 < KB2; ++kb2) {
    // ---- stage B tile (64 K x 16 pixels) into LDS as f16, im2col on the fly
    {
      int kg = kb2*64 + kl;
      int ci = kg / KSS;
      int r  = kg - ci*KSS;
      int dh = r / KS - PAD;
      int dw = (r - (r / KS)*KS) - PAD;
      const float* ip = in + (size_t)(bn*CIN + ci)*HW;
#pragma unroll
      for (int j = 0; j < 8; ++j) {
        int hh = ph[j] + dh, ww = pw[j] + dw;
        float val = (hh >= 0 && hh < FH && ww >= 0 && ww < FW) ? ip[hh*FW + ww] : 0.0f;
        Bs[(n0 + j)*66 + kl] = (_Float16)val;
      }
    }
    __syncthreads();

    const uint4* p0 = wpA0 + (size_t)kb2*128;
    const uint4* p1 = wpA1 + (size_t)kb2*128;
    if (kb2 + 1 < KB2) {                  // uniform branch; next step's A fragments
      __builtin_prefetch(p0 + 128, 0, 0);
      __builtin_prefetch(p1 + 128, 0, 0);
    }

    // ---- A fragments (pre-packed): 2 co tiles x 2 K halves, b128 loads
    Frag a00, a01, a10, a11, b0, b1;
    a00.q[0] = p0[0];   a00.q[1] = p0[1];
    a01.q[0] = p0[64];  a01.q[1] = p0[65];
    a10.q[0] = p1[0];   a10.q[1] = p1[1];
    a11.q[0] = p1[64];  a11.q[1] = p1[65];

    // ---- B fragments from LDS: lane<16 -> K {0..7,16..23}, lane>=16 -> K {8..15,24..31}
    const unsigned int* bsrc =
        (const unsigned int*)(&Bs[(lane & 15)*66 + ((lane & 16) >> 1)]);
#pragma unroll
    for (int v = 0; v < 8; ++v) b0.u[v] = bsrc[v + (v & 4)];        // K   0..31 half
#pragma unroll
    for (int v = 0; v < 8; ++v) b1.u[v] = bsrc[v + (v & 4) + 16];   // K  32..63 half

    acc[0] = __builtin_amdgcn_wmma_f32_16x16x32_f16(false, a00.v, false, b0.v,
                                                    (short)0, acc[0], false, false);
    acc[1] = __builtin_amdgcn_wmma_f32_16x16x32_f16(false, a10.v, false, b0.v,
                                                    (short)0, acc[1], false, false);
    acc[0] = __builtin_amdgcn_wmma_f32_16x16x32_f16(false, a01.v, false, b1.v,
                                                    (short)0, acc[0], false, false);
    acc[1] = __builtin_amdgcn_wmma_f32_16x16x32_f16(false, a11.v, false, b1.v,
                                                    (short)0, acc[1], false, false);
    __syncthreads();
  }

  // ---- epilogue (C/D layout: lanes 0-15 N=lane M=r, lanes 16-31 N=lane-16 M=r+8)
  const int nE  = lane & 15;
  const int hwE = hw0 + nE;
  const int PE  = P0 + nE;
#pragma unroll
  for (int t2 = 0; t2 < 2; ++t2) {
    int coBase = (coT0 + t2)*16;
#pragma unroll
    for (int r = 0; r < 8; ++r) {
      int col = (lane < 16) ? r : (r + 8);
      int co  = coBase + col;
      float y = acc[t2][r];
      if (MODE == 0) {
        y = y * scale[co] + bias[co];
        y = fmaxf(y, 0.0f);
        out[(size_t)(bn*COUT + co)*HW + hwE] = y;   // NCHW
      } else {
        out[(size_t)PE*COUT + co] = y + bias[co];   // NHWC (for coalesced splat)
      }
    }
  }
}

// ---------------------------------------------------------------------------
// Zero-fill
// ---------------------------------------------------------------------------
__global__ __launch_bounds__(256) void zero_kernel(float* __restrict__ p, int n) {
  int i = blockIdx.x*256 + threadIdx.x;
  if (i < n) p[i] = 0.0f;
}

// ---------------------------------------------------------------------------
// Fused lift + splat: one block per pixel, threads = channels.
// feat is NHWC so the 128-channel read is one coalesced line; BEV accumulator
// is [voxel][c] so the 128 atomics per (pixel,d) hit 2 contiguous cache lines.
// B=1 -> segment id == voxel id.
// ---------------------------------------------------------------------------
__global__ __launch_bounds__(128) void splat_kernel(
    const float* __restrict__ dist, const int* __restrict__ vox,
    const float* __restrict__ feat, float* __restrict__ bev) {
  int p = blockIdx.x;
  int c = threadIdx.x;
  int bn = p / HW, hw = p - bn*HW;
  float f = feat[(size_t)p*CFEAT + c];
  const float* dp = dist + (size_t)bn*NDEPTH*HW + hw;
  const int*   vp = vox  + (size_t)bn*NDEPTH*HW + hw;
  for (int d = 0; d < NDEPTH; ++d) {
    float pd = dp[(size_t)d*HW];      // broadcast across the block
    int   v  = vp[(size_t)d*HW];      // broadcast across the block
    atomicAdd(&bev[(size_t)v*CFEAT + c], pd * f);
  }
}

// ---------------------------------------------------------------------------
// BEV [voxel][c] -> output (B,C,X,Y) = [c][voxel]
// ---------------------------------------------------------------------------
__global__ __launch_bounds__(256) void bev_transpose(const float* __restrict__ bev,
                                                     float* __restrict__ out) {
  int i = blockIdx.x*256 + threadIdx.x;
  if (i >= XY*CFEAT) return;
  int c  = i >> 14;           // i / 16384
  int vx = i & (XY - 1);
  out[i] = bev[(size_t)vx*CFEAT + c];
}

// ---------------------------------------------------------------------------
extern "C" void kernel_launch(void* const* d_in, const int* in_sizes, int n_in,
                              void* d_out, int out_size, void* d_ws, size_t ws_size,
                              hipStream_t stream) {
  const float* rel = (const float*)d_in[0];
  const float* img = (const float*)d_in[1];
  const int*   vox = (const int*)d_in[2];
  const float* lsc = (const float*)d_in[3];
  const float* shf = (const float*)d_in[4];
  const float* lsg = (const float*)d_in[5];
  const float* w1  = (const float*)d_in[6];
  const float* c1b = (const float*)d_in[7];
  const float* g1  = (const float*)d_in[8];
  const float* bb1 = (const float*)d_in[9];
  const float* m1  = (const float*)d_in[10];
  const float* v1  = (const float*)d_in[11];
  const float* w2  = (const float*)d_in[12];
  const float* c2b = (const float*)d_in[13];
  const float* g2  = (const float*)d_in[14];
  const float* bb2 = (const float*)d_in[15];
  const float* m2  = (const float*)d_in[16];
  const float* v2  = (const float*)d_in[17];
  const float* w3  = (const float*)d_in[18];
  const float* c3b = (const float*)d_in[19];
  float* out = (float*)d_out;

  // ---- workspace carve-out (256B aligned regions), ~50 MB total
  char* ws = (char*)d_ws;
  auto carve = [&](size_t bytes) -> char* {
    char* p = ws;
    ws += (bytes + 255) & ~(size_t)255;
    return p;
  };
  const int NW1 = (256/16)*72*256;   // 294912 u32 fragments (conv1/conv2)
  const int NW3 = (128/16)*8*256;    // 16384 u32 fragments (conv3)
  unsigned int* w1p = (unsigned int*)carve((size_t)NW1*4);
  unsigned int* w2p = (unsigned int*)carve((size_t)NW1*4);
  unsigned int* w3p = (unsigned int*)carve((size_t)NW3*4);
  float* sb1  = (float*)carve(512*4);
  float* sb2  = (float*)carve(512*4);
  float* b3   = (float*)carve(128*4);
  float* h1   = (float*)carve((size_t)BNC*256*HW*4);
  float* h2   = (float*)carve((size_t)BNC*256*HW*4);
  float* dist = (float*)carve((size_t)BNC*NDEPTH*HW*4);
  float* bev  = (float*)carve((size_t)XY*CFEAT*4);
  float* feat = h1;   // conv1 activations dead after conv2; reuse for NHWC feat

  // ---- prep
  pack_weights<256,256,3><<<(NW1 + 255)/256, 256, 0, stream>>>(w1, w1p);
  pack_weights<256,256,3><<<(NW1 + 255)/256, 256, 0, stream>>>(w2, w2p);
  pack_weights<256,128,1><<<(NW3 + 255)/256, 256, 0, stream>>>(w3, w3p);
  prep_sb<<<3, 256, 0, stream>>>(c1b, g1, bb1, m1, v1,
                                 c2b, g2, bb2, m2, v2,
                                 c3b, sb1, sb2, b3);

  // ---- depth distribution
  depth_kernel<<<(NPIX + 255)/256, 256, 0, stream>>>(rel, lsc, shf, lsg, dist);

  // ---- conv stack (WMMA implicit GEMM)
  dim3 gc12(NPIX/16, 2);   // 1056 pixel tiles x (16 co-tiles / 8 per block)
  conv_wmma<256,256,3,1,0><<<gc12, 128, 0, stream>>>(img, w1p, sb1, sb1 + 256, h1);
  conv_wmma<256,256,3,1,0><<<gc12, 128, 0, stream>>>(h1,  w2p, sb2, sb2 + 256, h2);
  dim3 gc3(NPIX/16, 1);    // 8 co-tiles total
  conv_wmma<256,128,1,0,1><<<gc3, 128, 0, stream>>>(h2, w3p, b3, b3, feat);

  // ---- fused lift + splat into L2-resident BEV, then layout fix
  zero_kernel<<<(XY*CFEAT + 255)/256, 256, 0, stream>>>(bev, XY*CFEAT);
  splat_kernel<<<NPIX, 128, 0, stream>>>(dist, vox, feat, bev);
  bev_transpose<<<(XY*CFEAT + 255)/256, 256, 0, stream>>>(bev, out);
}
